// DotProductAttention_10075993277137
// MI455X (gfx1250) — matrix operand order repaired and checked by reading, not compiled, not run on hardware
//
#include <hip/hip_runtime.h>
#include <stdint.h>

typedef __attribute__((ext_vector_type(16))) __bf16 v16bf;
typedef __attribute__((ext_vector_type(8)))  __bf16 v8bf;
typedef __attribute__((ext_vector_type(8)))  float  v8f;

#define SEQ 256      // Q == K
#define DIM 32       // head dim
#define QTILE 16
#define KSTR 40      // K_lds row stride in bf16 (80 B; 16 rows hit distinct bank groups)
#define VTSTR 264    // Vt_lds row stride in bf16 (528 B; 16 rows -> 4-bank steps, conflict-free)
#define OSTR 33      // o-staging row stride in f32 (odd -> conflict-free scatter)
#define NEGV (-1000000.0f)

union BfPack { uint32_t u; __bf16 h[2]; };
union B2Cast { uint32_t u[8]; v16bf v; };

__global__ __launch_bounds__(256) void attn_swin_kernel(
    const float* __restrict__ qm, const float* __restrict__ km,
    const float* __restrict__ vm, const int* __restrict__ valid_lens,
    const float* __restrict__ wmask, float* __restrict__ out,
    int num_windows)
{
  __shared__ __align__(16) __bf16 k_lds[SEQ * KSTR];     // 20480 B, row-major keys
  __shared__ __align__(16) __bf16 vt_lds[DIM * VTSTR];   // 16896 B, transposed values
  __shared__ __align__(16) float  o_lds[8 * QTILE * OSTR]; // 16896 B, per-wave output stage

  const int b    = blockIdx.x;
  const int tid  = threadIdx.x;
  const int lane = tid & 31;
  const int wave = tid >> 5;
  const int hl   = lane >> 4;   // lane half (0/1)
  const int l16  = lane & 15;

  // ---- stage K (row-major bf16) and V^T into LDS: one row per thread ----
  {
    const float4* kr = (const float4*)(km + (size_t)b * SEQ * DIM + (size_t)tid * DIM);
    const float4* vr = (const float4*)(vm + (size_t)b * SEQ * DIM + (size_t)tid * DIM);
#pragma unroll
    for (int c = 0; c < 8; ++c) {
      float4 kf = kr[c];
      float4 vf = vr[c];
      __bf16* kd = &k_lds[tid * KSTR + 4 * c];
      kd[0] = (__bf16)kf.x; kd[1] = (__bf16)kf.y; kd[2] = (__bf16)kf.z; kd[3] = (__bf16)kf.w;
      vt_lds[(4 * c + 0) * VTSTR + tid] = (__bf16)vf.x;
      vt_lds[(4 * c + 1) * VTSTR + tid] = (__bf16)vf.y;
      vt_lds[(4 * c + 2) * VTSTR + tid] = (__bf16)vf.z;
      vt_lds[(4 * c + 3) * VTSTR + tid] = (__bf16)vf.w;
    }
  }
  __syncthreads();

  const int   len   = valid_lens[b];
  const float scale = 0.17677669529663687f; // 1/sqrt(32)
  const int   w_idx = (b >> 3) % num_windows; // (b / NUM_HEADS) % num_windows
  const float* wm   = wmask + (size_t)w_idx * SEQ * SEQ;

#pragma unroll
  for (int qt = wave; qt < SEQ / QTILE; qt += 8) {
    const int qbase = qt * QTILE;
    const int qrow  = qbase + l16;

    // ---- B1 = Q^T tile (32d x 16q): lane half picks d {0..7,16..23} / {8..15,24..31} ----
    v16bf b1;
    {
      const float* qp = qm + (size_t)b * SEQ * DIM + (size_t)qrow * DIM + hl * 8;
#pragma unroll
      for (int i = 0; i < 16; ++i) b1[i] = (__bf16)qp[i < 8 ? i : i + 8];
    }

    // ---- GEMM1: S^T tiles, D-layout: q = l16, key = kt*16 + hl*8 + j ----
    float s[16][8];
#pragma unroll
    for (int kt = 0; kt < 16; ++kt) {
      const __bf16* ar = &k_lds[(kt * QTILE + l16) * KSTR + hl * 8];
      v8bf alo = *(const v8bf*)ar;          // d chunk {0..7}/{8..15}
      v8bf ahi = *(const v8bf*)(ar + 16);   // d chunk {16..23}/{24..31}
      v16bf a1 = __builtin_shufflevector(alo, ahi,
          0,1,2,3,4,5,6,7,8,9,10,11,12,13,14,15);
      v8f c = {};
      c = __builtin_amdgcn_wmma_f32_16x16x32_bf16(false, a1, false, b1,
                                                  (short)0, c, false, false);
      const float* mp = wm + (size_t)qrow * SEQ + kt * QTILE + hl * 8;
      float4 m0 = *(const float4*)mp;
      float4 m1 = *(const float4*)(mp + 4);
      float mv[8] = {m0.x, m0.y, m0.z, m0.w, m1.x, m1.y, m1.z, m1.w};
      const int kb = kt * QTILE + hl * 8;
#pragma unroll
      for (int j = 0; j < 8; ++j)
        s[kt][j] = (kb + j < len) ? (c[j] * scale + mv[j]) : NEGV;
    }

    // ---- softmax over keys (per q = l16): lane-local then one xor16 ----
    float mx = -3.0e38f;
#pragma unroll
    for (int kt = 0; kt < 16; ++kt)
#pragma unroll
      for (int j = 0; j < 8; ++j) mx = fmaxf(mx, s[kt][j]);
    mx = fmaxf(mx, __shfl_xor(mx, 16, 32));

    float lsum = 0.0f;
    uint32_t pk[16][4]; // packed bf16 pairs of exp(s - mx)
#pragma unroll
    for (int kt = 0; kt < 16; ++kt)
#pragma unroll
      for (int i = 0; i < 4; ++i) {
        float e0 = __expf(s[kt][2 * i]     - mx);
        float e1 = __expf(s[kt][2 * i + 1] - mx);
        lsum += e0 + e1;
        BfPack p; p.h[0] = (__bf16)e0; p.h[1] = (__bf16)e1;
        pk[kt][i] = p.u;
      }
    lsum += __shfl_xor(lsum, 16, 32);
    const float inv = 1.0f / lsum;

    // ---- GEMM2: O^T = V^T x P^T ; P^T B-layout taken directly from this lane's pk ----
    v8f acc0 = {}; v8f acc1 = {};
#pragma unroll
    for (int tp = 0; tp < 8; ++tp) {
      B2Cast bc;
#pragma unroll
      for (int i = 0; i < 4; ++i) {
        uint32_t snd = hl ? pk[2 * tp][i] : pk[2 * tp + 1][i];
        uint32_t rcv = (uint32_t)__shfl_xor((int)snd, 16, 32);
        bc.u[i]     = pk[2 * tp][i];
        bc.u[4 + i] = pk[2 * tp + 1][i];
      }
      {
        const __bf16* vr = &vt_lds[(0 * 16 + l16) * VTSTR + tp * 32 + hl * 8];
        v8bf lo = *(const v8bf*)vr;
        v8bf hi = *(const v8bf*)(vr + 16);
        v16bf a2 = __builtin_shufflevector(lo, hi,
            0,1,2,3,4,5,6,7,8,9,10,11,12,13,14,15);
        acc0 = __builtin_amdgcn_wmma_f32_16x16x32_bf16(false, a2, false, bc.v,
                                                       (short)0, acc0, false, false);
      }
      {
        const __bf16* vr = &vt_lds[(1 * 16 + l16) * VTSTR + tp * 32 + hl * 8];
        v8bf lo = *(const v8bf*)vr;
        v8bf hi = *(const v8bf*)(vr + 16);
        v16bf a2 = __builtin_shufflevector(lo, hi,
            0,1,2,3,4,5,6,7,8,9,10,11,12,13,14,15);
        acc1 = __builtin_amdgcn_wmma_f32_16x16x32_bf16(false, a2, false, bc.v,
                                                       (short)0, acc1, false, false);
      }
    }

    // ---- normalize, stage through LDS, coalesced store ----
    float* ow = &o_lds[wave * QTILE * OSTR];
#pragma unroll
    for (int j = 0; j < 8; ++j) {
      ow[l16 * OSTR +      hl * 8 + j] = acc0[j] * inv; // d = 0..15
      ow[l16 * OSTR + 16 + hl * 8 + j] = acc1[j] * inv; // d = 16..31
    }
#pragma unroll
    for (int r = 0; r < QTILE; ++r) {
      out[(size_t)b * SEQ * DIM + (size_t)(qbase + r) * DIM + lane] = ow[r * OSTR + lane];
    }
  }
}

extern "C" void kernel_launch(void* const* d_in, const int* in_sizes, int n_in,
                              void* d_out, int out_size, void* d_ws, size_t ws_size,
                              hipStream_t stream) {
  const float* q  = (const float*)d_in[0];
  const float* k  = (const float*)d_in[1];
  const float* v  = (const float*)d_in[2];
  const int*   vl = (const int*)d_in[3];
  const float* wm = (const float*)d_in[4];
  float* out = (float*)d_out;

  const int n = in_sizes[3];                       // batch count (valid_lens is [n])
  const int num_windows = in_sizes[4] / (SEQ * SEQ);

  attn_swin_kernel<<<n, 256, 0, stream>>>(q, k, v, vl, wm, out, num_windows);
}
